// TurboQuantizedAttention_19550691131801
// MI455X (gfx1250) — matrix operand order repaired and checked by reading, not compiled, not run on hardware
//
#include <hip/hip_runtime.h>
#include <hip/hip_bf16.h>
#include <cstdint>
#include <cstddef>

// ---------------------------------------------------------------------------
// Problem constants (from reference): B=2, S=1024, H=4096, NH=32, NKV=8, HD=128
// ---------------------------------------------------------------------------
#define Bc    2
#define Sc    1024
#define Hc    4096
#define NHc   32
#define NKVc  8
#define HDc   128

typedef __attribute__((ext_vector_type(16))) __bf16 v16bf;
typedef __attribute__((ext_vector_type(8)))  __bf16 v8bf;
typedef __attribute__((ext_vector_type(8)))  float  v8f;

static __device__ __forceinline__ unsigned short f2bf(float f) {
  unsigned int u = __float_as_uint(f);
  u += 0x7FFFu + ((u >> 16) & 1u);           // round-to-nearest-even
  return (unsigned short)(u >> 16);
}

// Assemble a 16-half A/B fragment from two contiguous 8-half (16B) chunks.
static __device__ __forceinline__ v16bf cat16(v8bf lo, v8bf hi) {
  v16bf r;
#pragma unroll
  for (int i = 0; i < 8; ++i) { r[i] = lo[i]; r[i + 8] = hi[i]; }
  return r;
}

// A-matrix (16-bit, 16x32) per-lane load. Caller passes base already offset by
// (k0 + hi*8) halves; element i<8 at +0, i>=8 at +16 halves (ISA 7.12.2 layout).
static __device__ __forceinline__ v16bf ldA(const unsigned short* p) {
  v8bf lo = *(const v8bf*)(p);
  v8bf hi = *(const v8bf*)(p + 16);
  return cat16(lo, hi);
}

// ---------------------------------------------------------------------------
// Kernel 1: f32 -> bf16 conversion of hidden states
// ---------------------------------------------------------------------------
__global__ void f32_to_bf16_kernel(const float* __restrict__ in,
                                   unsigned short* __restrict__ out, int n) {
  int i = (blockIdx.x * blockDim.x + threadIdx.x) * 4;
  if (i + 3 < n) {
    float4 v = *(const float4*)(in + i);
    out[i + 0] = f2bf(v.x);
    out[i + 1] = f2bf(v.y);
    out[i + 2] = f2bf(v.z);
    out[i + 3] = f2bf(v.w);
  }
}

// ---------------------------------------------------------------------------
// Kernel 2: generic dequant-GEMM, Y = X(bf16 [M,K]) * W^T(int8-in-int32 [N,K])
//           * scale[N] (+ bias[N]).  BM=128, BN=128, BK=32, 256 thr = 8 waves.
// Each wave owns a 32x64 output slab = 2x4 fragments of 16x16.
// mode 0: f32 [M,N]       (O projection -> d_out)
// mode 1: bf16 [B,h,S,HD] (Q, K projections, head-split)
// mode 2: bf16 [B,h,HD,S] (V projection, head-transposed for flash PV)
// ---------------------------------------------------------------------------
#define BM 128
#define BN 128
#define BK 32

__global__ __launch_bounds__(256)
void gemm_bf16_wmma(const unsigned short* __restrict__ X,
                    const int*   __restrict__ W,
                    const float* __restrict__ scale,
                    const float* __restrict__ bias,
                    void* __restrict__ Y,
                    int M, int N, int K,
                    int mode, int nheads) {
  __shared__ alignas(128) unsigned short Wt[BN * BK];   // dequantized B tile, [n][k]

  const int tid   = threadIdx.x;
  const int lane  = tid & 31;
  const int wave  = tid >> 5;
  const int wm    = wave & 3;      // 0..3 : 32-row slab
  const int wn    = wave >> 2;     // 0..1 : 64-col slab
  const int laneN = lane & 15;
  const int hi    = lane >> 4;     // lane half (ISA A/B/C layouts key off this)

  const int mBase = blockIdx.x * BM;
  const int nBase = blockIdx.y * BN;

  // Weight staging ownership: thread -> row nLoc, k-halves kLoc..kLoc+15
  const int nLoc = tid >> 1;
  const int kLoc = (tid & 1) * 16;
  const float wsc = scale[nBase + nLoc];
  const int* wsrc = W + (size_t)(nBase + nLoc) * K + kLoc;

  v8f acc[2][4];
#pragma unroll
  for (int a = 0; a < 2; ++a)
#pragma unroll
    for (int b = 0; b < 4; ++b)
#pragma unroll
      for (int e = 0; e < 8; ++e) acc[a][b][e] = 0.0f;

  const unsigned short* arow0 =
      X + (size_t)(mBase + wm * 32 + laneN) * K + hi * 8;
  const unsigned short* arow1 =
      X + (size_t)(mBase + wm * 32 + 16 + laneN) * K + hi * 8;

  for (int k0 = 0; k0 < K; k0 += BK) {
    // software prefetch of the next weight chunk (global_prefetch_b8)
    if (k0 + BK < K) __builtin_prefetch(wsrc + k0 + BK, 0, 1);

    // ---- stage + dequantize weights into LDS (int8-in-int32 -> bf16*scale)
    int wv16[16];
    *(int4*)(wv16 + 0)  = *(const int4*)(wsrc + k0 + 0);
    *(int4*)(wv16 + 4)  = *(const int4*)(wsrc + k0 + 4);
    *(int4*)(wv16 + 8)  = *(const int4*)(wsrc + k0 + 8);
    *(int4*)(wv16 + 12) = *(const int4*)(wsrc + k0 + 12);
    unsigned short* dst = Wt + nLoc * BK + kLoc;
#pragma unroll
    for (int i = 0; i < 16; ++i) dst[i] = f2bf((float)wv16[i] * wsc);
    __syncthreads();

    // ---- A fragments straight from global bf16 (L2-resident)
    v16bf a0 = ldA(arow0 + k0);
    v16bf a1 = ldA(arow1 + k0);

    // ---- 8 WMMAs per wave per k-step
#pragma unroll
    for (int fn = 0; fn < 4; ++fn) {
      const unsigned short* bp =
          Wt + (wn * 64 + fn * 16 + laneN) * BK + hi * 8;
      v16bf bb = ldA(bp);   // two 8-half chunks at +0 and +16 (2x ds_load_b128)
      acc[0][fn] = __builtin_amdgcn_wmma_f32_16x16x32_bf16(
          false, a0, false, bb, (short)0, acc[0][fn], false, false);
      acc[1][fn] = __builtin_amdgcn_wmma_f32_16x16x32_bf16(
          false, a1, false, bb, (short)0, acc[1][fn], false, false);
    }
    __syncthreads();
  }

  // ---- epilogue: C/D layout is N=lane%16, M=e + 8*hi (ISA 7.12.2)
#pragma unroll
  for (int fm = 0; fm < 2; ++fm) {
#pragma unroll
    for (int fn = 0; fn < 4; ++fn) {
      const int col = nBase + wn * 64 + fn * 16 + laneN;
      const float bv = bias ? bias[col] : 0.0f;
#pragma unroll
      for (int e = 0; e < 8; ++e) {
        const int row = mBase + wm * 32 + fm * 16 + e + hi * 8;
        const float val = acc[fm][fn][e] + bv;
        if (mode == 0) {
          ((float*)Y)[(size_t)row * N + col] = val;
        } else {
          const int bi = row >> 10;          // / S (1024)
          const int s  = row & (Sc - 1);
          const int h  = col >> 7;           // / HD (128)
          const int hd = col & (HDc - 1);
          size_t idx;
          if (mode == 1)
            idx = (((size_t)(bi * nheads + h)) * Sc + s) * HDc + hd;
          else
            idx = (((size_t)(bi * nheads + h)) * HDc + hd) * Sc + s;
          ((unsigned short*)Y)[idx] = f2bf(val);
        }
      }
    }
  }
}

// ---------------------------------------------------------------------------
// Kernel 3: causal GQA flash attention, 1 WG per (b, head, 64-row q-tile),
// 4 waves x 16 query rows, K/V tiles of 64, all matmuls bf16 WMMA.
// ---------------------------------------------------------------------------
__global__ __launch_bounds__(128)
void flash_attn_wmma(const unsigned short* __restrict__ Qm,   // [B,NH,S,HD] bf16
                     const unsigned short* __restrict__ Km,   // [B,NKV,S,HD] bf16
                     const unsigned short* __restrict__ Vm,   // [B,NKV,HD,S] bf16 (transposed)
                     const float* __restrict__ mask,          // [B,1,S,S] f32
                     unsigned short* __restrict__ CTX) {      // [B,S,NH,HD] bf16
  __shared__ alignas(128) unsigned short Kl[64 * HDc];        // [kv][hd]  16 KB
  __shared__ alignas(128) unsigned short Vl[HDc * 64];        // [hd][kv]  16 KB
  __shared__ alignas(128) unsigned short Pl[4 * 16 * 64];     // per-wave P  8 KB

  const int tid   = threadIdx.x;
  const int lane  = tid & 31;
  const int wave  = tid >> 5;     // 0..3
  const int laneN = lane & 15;
  const int hi    = lane >> 4;

  const int qt  = blockIdx.x;     // q tile (0..15)
  const int h   = blockIdx.y;     // head (0..31)
  const int bb  = blockIdx.z;     // batch
  const int kvh = h >> 2;         // GQA: GROUPS = 4

  // Q fragments live in registers across the whole kv loop (4 x 16x32)
  const int qrow = qt * 64 + wave * 16 + laneN;
  const unsigned short* qbase =
      Qm + (((size_t)(bb * NHc + h)) * Sc + qrow) * HDc + hi * 8;
  v16bf aq[4];
#pragma unroll
  for (int kc = 0; kc < 4; ++kc) aq[kc] = ldA(qbase + kc * 32);

  float mi[8], li[8];
  v8f o[8];
#pragma unroll
  for (int e = 0; e < 8; ++e) { mi[e] = -1e30f; li[e] = 0.0f; }
#pragma unroll
  for (int f = 0; f < 8; ++f)
#pragma unroll
    for (int e = 0; e < 8; ++e) o[f][e] = 0.0f;

  const float sm_scale = 0.08838834764831845f;  // 1/sqrt(128)
  const unsigned short* kgbase = Km + ((size_t)(bb * NKVc + kvh)) * Sc * HDc;
  const unsigned short* vgbase = Vm + ((size_t)(bb * NKVc + kvh)) * HDc * Sc;

  for (int kt = 0; kt <= qt; ++kt) {          // causal: skip upper tiles
    const int kv0 = kt * 64;

    // ---- stage K tile [64][128] and V^T tile [128][64] into LDS (b128 copies)
    {
      const int r = tid >> 1, part = tid & 1;
      const uint4* ksrc =
          (const uint4*)(kgbase + (size_t)(kv0 + r) * HDc + part * 64);
      uint4* kdst = (uint4*)(Kl + r * HDc + part * 64);
#pragma unroll
      for (int j = 0; j < 8; ++j) kdst[j] = ksrc[j];

      const uint4* vsrc = (const uint4*)(vgbase + (size_t)tid * Sc + kv0);
      uint4* vdst = (uint4*)(Vl + tid * 64);
#pragma unroll
      for (int j = 0; j < 8; ++j) vdst[j] = vsrc[j];
    }
    __syncthreads();

    // ---- scores S = Q K^T  (4 n-frags x 4 k-chunks = 16 WMMAs)
    v8f sc[4];
#pragma unroll
    for (int fn = 0; fn < 4; ++fn)
#pragma unroll
      for (int e = 0; e < 8; ++e) sc[fn][e] = 0.0f;
#pragma unroll
    for (int fn = 0; fn < 4; ++fn) {
#pragma unroll
      for (int kc = 0; kc < 4; ++kc) {
        v16bf bk = ldA(Kl + (fn * 16 + laneN) * HDc +
                                   kc * 32 + hi * 8);
        sc[fn] = __builtin_amdgcn_wmma_f32_16x16x32_bf16(
            false, aq[kc], false, bk, (short)0, sc[fn], false, false);
      }
    }

    // ---- scale; additive mask only on the diagonal tile (0 below diagonal)
    if (kt == qt) {
#pragma unroll
      for (int fn = 0; fn < 4; ++fn) {
        const int col = kv0 + fn * 16 + laneN;
#pragma unroll
        for (int e = 0; e < 8; ++e) {
          const int row = qt * 64 + wave * 16 + e + hi * 8;
          sc[fn][e] = sc[fn][e] * sm_scale +
                      mask[((size_t)bb * Sc + row) * Sc + col];
        }
      }
    } else {
#pragma unroll
      for (int fn = 0; fn < 4; ++fn)
#pragma unroll
        for (int e = 0; e < 8; ++e) sc[fn][e] *= sm_scale;
    }

    // ---- online softmax: row reductions via shfl_xor within 16-lane groups
    float mnew[8], alpha[8];
#pragma unroll
    for (int e = 0; e < 8; ++e) {
      float r = fmaxf(fmaxf(sc[0][e], sc[1][e]), fmaxf(sc[2][e], sc[3][e]));
      r = fmaxf(r, __shfl_xor(r, 1));
      r = fmaxf(r, __shfl_xor(r, 2));
      r = fmaxf(r, __shfl_xor(r, 4));
      r = fmaxf(r, __shfl_xor(r, 8));
      mnew[e]  = fmaxf(mi[e], r);
      alpha[e] = __expf(mi[e] - mnew[e]);
      mi[e]    = mnew[e];
    }
#pragma unroll
    for (int fn = 0; fn < 4; ++fn)
#pragma unroll
      for (int e = 0; e < 8; ++e) sc[fn][e] = __expf(sc[fn][e] - mnew[e]);
#pragma unroll
    for (int e = 0; e < 8; ++e) {
      float r = sc[0][e] + sc[1][e] + sc[2][e] + sc[3][e];
      r += __shfl_xor(r, 1);
      r += __shfl_xor(r, 2);
      r += __shfl_xor(r, 4);
      r += __shfl_xor(r, 8);
      li[e] = li[e] * alpha[e] + r;
    }
#pragma unroll
    for (int f = 0; f < 8; ++f)
#pragma unroll
      for (int e = 0; e < 8; ++e) o[f][e] *= alpha[e];

    // ---- transpose P (C-layout -> A-layout) through per-wave LDS scratch;
    //      same-wave DS ops are in-order (ISA 7.3), so no barrier needed.
    unsigned short* pw = Pl + wave * (16 * 64);
#pragma unroll
    for (int fn = 0; fn < 4; ++fn)
#pragma unroll
      for (int e = 0; e < 8; ++e)
        pw[(e + hi * 8) * 64 + fn * 16 + laneN] = f2bf(sc[fn][e]);

    // ---- O += P V  (8 hd-frags x 2 k-chunks = 16 WMMAs)
#pragma unroll
    for (int kc = 0; kc < 2; ++kc) {
      v16bf ap = ldA(pw + laneN * 64 + kc * 32 + hi * 8);
#pragma unroll
      for (int f = 0; f < 8; ++f) {
        v16bf bv = ldA(Vl + (f * 16 + laneN) * 64 +
                                   kc * 32 + hi * 8);
        o[f] = __builtin_amdgcn_wmma_f32_16x16x32_bf16(
            false, ap, false, bv, (short)0, o[f], false, false);
      }
    }
    __syncthreads();   // protect K/V tiles before next staging pass
  }

  // ---- normalize and store context as bf16 [B,S,NH,HD] (O-proj activation)
#pragma unroll
  for (int f = 0; f < 8; ++f) {
    const int hd = f * 16 + laneN;
#pragma unroll
    for (int e = 0; e < 8; ++e) {
      const int row = qt * 64 + wave * 16 + e + hi * 8;
      const float val = o[f][e] / li[e];
      CTX[(((size_t)bb * Sc + row) * NHc + h) * HDc + hd] = f2bf(val);
    }
  }
}

// ---------------------------------------------------------------------------
// Host-side orchestration
// ---------------------------------------------------------------------------
extern "C" void kernel_launch(void* const* d_in, const int* in_sizes, int n_in,
                              void* d_out, int out_size, void* d_ws,
                              size_t ws_size, hipStream_t stream) {
  (void)in_sizes; (void)n_in; (void)out_size; (void)ws_size;

  const float* hidden = (const float*)d_in[0];
  const float* mask   = (const float*)d_in[1];
  const int*   wq     = (const int*)d_in[2];
  const float* wq_s   = (const float*)d_in[3];
  const float* bq     = (const float*)d_in[4];
  const int*   wk     = (const int*)d_in[5];
  const float* wk_s   = (const float*)d_in[6];
  const float* bk     = (const float*)d_in[7];
  const int*   wv     = (const int*)d_in[8];
  const float* wv_s   = (const float*)d_in[9];
  const int*   wo     = (const int*)d_in[10];
  const float* wo_s   = (const float*)d_in[11];

  const size_t M = (size_t)Bc * Sc;  // 2048

  // bf16 workspace layout (~59 MB total)
  unsigned short* Xb  = (unsigned short*)d_ws;                 // [M, H]
  unsigned short* Qb  = Xb  + M * Hc;                          // [B,NH,S,HD]
  unsigned short* Kb  = Qb  + M * Hc;                          // [B,NKV,S,HD]
  unsigned short* Vtb = Kb  + (size_t)Bc * NKVc * Sc * HDc;    // [B,NKV,HD,S]
  unsigned short* Cb  = Vtb + (size_t)Bc * NKVc * Sc * HDc;    // [B,S,NH,HD]

  const int n = (int)(M * Hc);
  f32_to_bf16_kernel<<<(n / 4 + 255) / 256, 256, 0, stream>>>(hidden, Xb, n);

  // Q/K/V projections (dequant int8 weights on the fly)
  gemm_bf16_wmma<<<dim3(16, 32), 256, 0, stream>>>(
      Xb, wq, wq_s, bq, Qb, 2048, 4096, 4096, /*mode=*/1, NHc);
  gemm_bf16_wmma<<<dim3(16, 8), 256, 0, stream>>>(
      Xb, wk, wk_s, bk, Kb, 2048, 1024, 4096, /*mode=*/1, NKVc);
  gemm_bf16_wmma<<<dim3(16, 8), 256, 0, stream>>>(
      Xb, wv, wv_s, nullptr, Vtb, 2048, 1024, 4096, /*mode=*/2, NKVc);

  // causal GQA flash attention
  flash_attn_wmma<<<dim3(Sc / 64, NHc, Bc), 128, 0, stream>>>(
      Qb, Kb, Vtb, mask, Cb);

  // output projection straight into d_out (f32)
  gemm_bf16_wmma<<<dim3(16, 32), 256, 0, stream>>>(
      Cb, wo, wo_s, nullptr, (float*)d_out, 2048, 4096, 4096, /*mode=*/0, 0);
}
